// TA_LSTMCell_7344394076790
// MI455X (gfx1250) — matrix operand order repaired and checked by reading, not compiled, not run on hardware
//
#include <hip/hip_runtime.h>
#include <hip/hip_bf16.h>
#include <stdint.h>

// ---------------------------------------------------------------------------
// TA-LSTM cell fused kernel for gfx1250 (MI455X), bf16 WMMA path.
//   gates = [input|hx] @ [Wih^T; Whh^T]  (K=512, N=1024)
//   C_ST  = tanh(cx @ W_decomp + b_decomp)
//   elementwise LSTM + time-decay epilogue, all in-register per wave.
// ---------------------------------------------------------------------------

typedef __attribute__((ext_vector_type(16))) __bf16 v16bf;
typedef __attribute__((ext_vector_type(8)))  float  v8f;

#define BB   65536
#define HHH  256
#define BH   (BB * HHH)

#define NT_G 64            // gate N tiles  (1024/16)
#define KS_G 16            // gate K steps  (512/32)
#define NT_D 16            // decomp N tiles (256/16)
#define KS_D 8             // decomp K steps (256/32)
#define FRAG_ELEMS 512     // 32 lanes * 16 bf16 per WMMA B fragment
#define WS_GATE_ELEMS (NT_G * KS_G * FRAG_ELEMS)
#define WS_DEC_ELEMS  (NT_D * KS_D * FRAG_ELEMS)

__device__ __forceinline__ unsigned short f2bf(float f) {
  // round-to-nearest-even fp32 -> bf16
  unsigned int u = __builtin_bit_cast(unsigned int, f);
  u += 0x7FFFu + ((u >> 16) & 1u);
  return (unsigned short)(u >> 16);
}

union FragU { uint4 q[2]; v16bf v; };

__device__ __forceinline__ float sigmoidf_(float x) {
  return 1.0f / (1.0f + __expf(-x));
}

// ---------------------------------------------------------------------------
// Prep: convert weights to bf16 and swizzle into per-lane WMMA B fragments.
// B fragment (32x16, K x N) lane layout, contiguous k order as in load_a:
// lane L holds column n = L%16; lanes 0-15 hold K 0..15,
// lanes 16-31 hold K 16..31 (v0.lo=K+0 .. v7.hi=K+15).
// ---------------------------------------------------------------------------
__global__ void ta_lstm_prep(const float* __restrict__ wih,
                             const float* __restrict__ whh,
                             const float* __restrict__ wdec,
                             unsigned short* __restrict__ ws) {
  int idx = blockIdx.x * blockDim.x + threadIdx.x;
  if (idx >= WS_GATE_ELEMS + WS_DEC_ELEMS) return;
  float v;
  if (idx < WS_GATE_ELEMS) {
    int frag   = idx / FRAG_ELEMS;
    int within = idx % FRAG_ELEMS;
    int lane = within >> 4, e = within & 15;
    int nt = frag / KS_G, ks = frag % KS_G;
    int n = nt * 16 + (lane & 15);
    int k = ks * 32 + ((lane >> 4) << 4) + e;        // 0..511
    // Wcat[k][n]: k<256 -> Wih^T, else Whh^T
    v = (k < 256) ? wih[n * 256 + k] : whh[n * 256 + (k - 256)];
  } else {
    int idx2   = idx - WS_GATE_ELEMS;
    int frag   = idx2 / FRAG_ELEMS;
    int within = idx2 % FRAG_ELEMS;
    int lane = within >> 4, e = within & 15;
    int nt = frag / KS_D, ks = frag % KS_D;
    int n = nt * 16 + (lane & 15);
    int k = ks * 32 + ((lane >> 4) << 4) + e;        // 0..255
    v = wdec[k * 256 + n];                            // W_decomp is [K][N]
  }
  ws[idx] = f2bf(v);
}

// ---------------------------------------------------------------------------
// A fragment (16x32 bf16, MxK) from row-major LDS panel.
// Contiguous k order: lanes 0-15 (m=lane): K {kb..kb+15};
//                    lanes 16-31 (m=lane-16): K {kb+16..kb+31}.
// ---------------------------------------------------------------------------
__device__ __forceinline__ v16bf load_a(const unsigned short* s, int rstride,
                                        int lane, int kbase) {
  int m     = lane & 15;
  int half8 = (lane >> 4) << 4;                       // 0 or 16
  const unsigned short* base = s + m * rstride + kbase + half8;
  FragU f;
  f.q[0] = *(const uint4*)(base);                     // K +0..7
  f.q[1] = *(const uint4*)(base + 8);                 // K +8..15
  return f.v;
}

__device__ __forceinline__ v16bf load_b(const unsigned short* g, int fragIdx,
                                        int lane) {
  const unsigned short* base = g + fragIdx * FRAG_ELEMS + lane * 16;
  FragU f;
  f.q[0] = ((const uint4*)base)[0];
  f.q[1] = ((const uint4*)base)[1];
  return f.v;
}

// ---------------------------------------------------------------------------
// Main fused kernel: 1 block = 16 batch rows, 8 waves.
// Wave w owns gate columns h in [w*32, w*32+32): N tiles {q*16 + 2w + j},
// q=0..3 (i,f,g,o), j=0..1, plus decomp tiles {2w, 2w+1} -> epilogue is
// fully in-register, no cross-wave gate exchange.
// ---------------------------------------------------------------------------
__global__ void __launch_bounds__(256, 1)
ta_lstm_main(const float* __restrict__ input, const float* __restrict__ t,
             const float* __restrict__ hx,    const float* __restrict__ cx,
             const float* __restrict__ bias_ih, const float* __restrict__ bias_hh,
             const float* __restrict__ b_decomp,
             const unsigned short* __restrict__ wsg,
             const unsigned short* __restrict__ wsd,
             float* __restrict__ out) {
  __shared__ unsigned short sA[16 * 512];   // [input|hx] bf16 panel, 16KB
  __shared__ unsigned short sCb[16 * 256];  // cx bf16 panel, 8KB
  __shared__ float          sCx[16 * 256];  // cx f32 for epilogue, 16KB
  __shared__ float          sTi[16];        // 1/t per row

  const int tid  = threadIdx.x;
  const int row0 = blockIdx.x * 16;

  // Stage & convert activations (fp32 -> bf16) into LDS.
  for (int e = tid; e < 16 * 256; e += 256) {
    int m = e >> 8, k = e & 255;
    long gi = (long)(row0 + m) * 256 + k;
    float vi = input[gi];
    float vh = hx[gi];
    float vc = cx[gi];
    sA[m * 512 + k]       = f2bf(vi);
    sA[m * 512 + 256 + k] = f2bf(vh);
    sCb[m * 256 + k]      = f2bf(vc);
    sCx[m * 256 + k]      = vc;
  }
  if (tid < 16) {
    float tv = t[row0 + tid];
    sTi[tid] = (tv != 0.0f) ? (1.0f / tv) : 0.0f;
  }
  __syncthreads();

  const int lane = tid & 31;
  const int w    = tid >> 5;

  const v8f zero = {0.f, 0.f, 0.f, 0.f, 0.f, 0.f, 0.f, 0.f};
  v8f accG[4][2];
  v8f accD[2];
#pragma unroll
  for (int q = 0; q < 4; ++q)
#pragma unroll
    for (int j = 0; j < 2; ++j) accG[q][j] = zero;
  accD[0] = zero; accD[1] = zero;

  // ---- gates GEMM: K = 512 (input|hx), 8 N tiles per wave ----
#pragma unroll 1
  for (int ks = 0; ks < KS_G; ++ks) {
    v16bf a = load_a(sA, 512, lane, ks * 32);
#pragma unroll
    for (int q = 0; q < 4; ++q) {
#pragma unroll
      for (int j = 0; j < 2; ++j) {
        int nt  = q * 16 + 2 * w + j;
        v16bf b = load_b(wsg, nt * KS_G + ks, lane);
        accG[q][j] = __builtin_amdgcn_wmma_f32_16x16x32_bf16(
            false, a, false, b, (short)0, accG[q][j], false, false);
      }
    }
  }

  // ---- decomp GEMM: C_pre = cx @ W_decomp, K = 256 ----
#pragma unroll 1
  for (int ks = 0; ks < KS_D; ++ks) {
    v16bf a = load_a(sCb, 256, lane, ks * 32);
#pragma unroll
    for (int j = 0; j < 2; ++j) {
      int nt  = 2 * w + j;
      v16bf b = load_b(wsd, nt * KS_D + ks, lane);
      accD[j] = __builtin_amdgcn_wmma_f32_16x16x32_bf16(
          false, a, false, b, (short)0, accD[j], false, false);
    }
  }

  // ---- epilogue, fully in-register per wave ----
  // C/D layout: VGPR r, lanes 0-15 -> M=r, lanes 16-31 -> M=8+r; N = lane%16.
  const int mbase = (lane >> 4) << 3;
#pragma unroll
  for (int j = 0; j < 2; ++j) {
    int h  = (2 * w + j) * 16 + (lane & 15);
    float bI = bias_ih[h]       + bias_hh[h];
    float bF = bias_ih[256 + h] + bias_hh[256 + h];
    float bG = bias_ih[512 + h] + bias_hh[512 + h];
    float bO = bias_ih[768 + h] + bias_hh[768 + h];
    float bd = b_decomp[h];
#pragma unroll
    for (int r = 0; r < 8; ++r) {
      int m = mbase + r;
      float ig  = sigmoidf_(accG[0][j][r] + bI);
      float fg  = sigmoidf_(accG[1][j][r] + bF);
      float gg  = tanhf(accG[2][j][r] + bG);
      float og  = sigmoidf_(accG[3][j][r] + bO);
      float cst = tanhf(accD[j][r] + bd);
      float cxv = sCx[m * 256 + h];
      float Ti  = sTi[m];
      float cadj = cxv - cst + Ti * cst;
      float cyv  = fg * cadj + ig * gg;
      float hyv  = og * tanhf(cyv);
      long o = (long)(row0 + m) * 256 + h;
      out[o]             = hyv;        // hy
      out[(long)BH + o]  = cyv;        // cy
    }
  }
}

// ---------------------------------------------------------------------------
extern "C" void kernel_launch(void* const* d_in, const int* in_sizes, int n_in,
                              void* d_out, int out_size, void* d_ws, size_t ws_size,
                              hipStream_t stream) {
  (void)in_sizes; (void)n_in; (void)out_size; (void)ws_size;
  const float* input = (const float*)d_in[0];
  const float* t     = (const float*)d_in[1];
  const float* hx    = (const float*)d_in[2];
  const float* cx    = (const float*)d_in[3];
  const float* wih   = (const float*)d_in[4];
  const float* whh   = (const float*)d_in[5];
  const float* bih   = (const float*)d_in[6];
  const float* bhh   = (const float*)d_in[7];
  const float* wdec  = (const float*)d_in[8];
  const float* bdec  = (const float*)d_in[9];
  float* out = (float*)d_out;

  unsigned short* wsg = (unsigned short*)d_ws;
  unsigned short* wsd = wsg + WS_GATE_ELEMS;

  int prep_total  = WS_GATE_ELEMS + WS_DEC_ELEMS;
  int prep_blocks = (prep_total + 255) / 256;
  ta_lstm_prep<<<prep_blocks, 256, 0, stream>>>(wih, whh, wdec, wsg);

  ta_lstm_main<<<BB / 16, 256, 0, stream>>>(input, t, hx, cx, bih, bhh, bdec,
                                            wsg, wsd, out);
}
